// TorchGroupGemmReduceRS_61168924229673
// MI455X (gfx1250) — compile-verified, matrix operand order checked by reading, not run on hardware
//
#include <hip/hip_runtime.h>
#include <hip/hip_bf16.h>

// ---------------- problem constants (match reference) ----------------
#define NUM_EXPERTS 8
#define TOPK        2
#define HIDDEN      2048
#define INTER       1024
#define NTOKENS     8192
#define NROWS       (NTOKENS * TOPK)   // 16384
#define SEG         16384              // per-expert rowmap segment capacity

// GEMM tiling
#define BM 32                          // rows per block
#define BN 256                         // cols per block
#define KB 32                          // K per WMMA step (bf16)
#define NK (INTER / KB)                // 32 K-steps
#define TILES_PER_SEG (SEG / BM)       // 512

typedef __attribute__((ext_vector_type(16))) __bf16       v16bf;
typedef __attribute__((ext_vector_type(8)))  float        v8f;
typedef __attribute__((ext_vector_type(4)))  unsigned int u32x4;

__device__ __forceinline__ unsigned f2bf1(float f) {
    // round-to-nearest-even f32 -> bf16 (result in low 16 bits)
    unsigned u = __float_as_uint(f);
    u += 0x7FFFu + ((u >> 16) & 1u);
    return u >> 16;
}
__device__ __forceinline__ unsigned pk2(float lo, float hi) {
    return f2bf1(lo) | (f2bf1(hi) << 16);
}

// ---------------------------------------------------------------------
// Kernel 1: x  (f32 [NROWS,INTER]) -> xb (bf16, same layout)
// ---------------------------------------------------------------------
__global__ __launch_bounds__(256) void xconv(const float* __restrict__ x,
                                             unsigned short* __restrict__ xb) {
    size_t i = ((size_t)blockIdx.x * 256 + threadIdx.x) * 8;
    float4 a = *(const float4*)(x + i);
    float4 b = *(const float4*)(x + i + 4);
    u32x4 o;
    o.x = pk2(a.x, a.y); o.y = pk2(a.z, a.w);
    o.z = pk2(b.x, b.y); o.w = pk2(b.z, b.w);
    *(u32x4*)(xb + i) = o;
}

// ---------------------------------------------------------------------
// Kernel 2: w (f32 [E,INTER,HIDDEN]) -> wbT (bf16 [E,HIDDEN,INTER])
// 64x64 tile transpose through LDS; coalesced read and write.
// ---------------------------------------------------------------------
__global__ __launch_bounds__(256) void wtrans(const float* __restrict__ w,
                                              unsigned short* __restrict__ wbT) {
    const int e = blockIdx.z, k0 = blockIdx.y * 64, h0 = blockIdx.x * 64;
    __shared__ alignas(16) unsigned short sT[64 * 72];   // [h][k], 144B row pitch
    const int tid = threadIdx.x;
    const float* wp = w + ((size_t)e * INTER + k0) * HIDDEN + h0;
    #pragma unroll
    for (int q = 0; q < 4; ++q) {                        // 1024 float4 reads
        int idx = q * 256 + tid;
        int kk  = idx >> 4;
        int h4  = (idx & 15) * 4;
        float4 v = *(const float4*)(wp + (size_t)kk * HIDDEN + h4);
        sT[(h4 + 0) * 72 + kk] = (unsigned short)f2bf1(v.x);
        sT[(h4 + 1) * 72 + kk] = (unsigned short)f2bf1(v.y);
        sT[(h4 + 2) * 72 + kk] = (unsigned short)f2bf1(v.z);
        sT[(h4 + 3) * 72 + kk] = (unsigned short)f2bf1(v.w);
    }
    __syncthreads();
    unsigned short* op = wbT + ((size_t)e * HIDDEN + h0) * INTER + k0;
    #pragma unroll
    for (int q = 0; q < 2; ++q) {                        // 512 x 16B writes
        int c = q * 256 + tid;
        int h = c >> 3;
        int kq = (c & 7) * 8;
        u32x4 d = *(const u32x4*)&sT[h * 72 + kq];
        *(u32x4*)(op + (size_t)h * INTER + kq) = d;
    }
}

// ---------------------------------------------------------------------
// Kernel 3: softmax + top-2 routing, scatter rows into per-expert bins
// ---------------------------------------------------------------------
__global__ __launch_bounds__(256) void moe_route(
    const float* __restrict__ logits, float* __restrict__ tw,
    int* __restrict__ rowmap, int* __restrict__ cursor) {
    int t = blockIdx.x * blockDim.x + threadIdx.x;
    if (t >= NTOKENS) return;

    float l[NUM_EXPERTS];
    #pragma unroll
    for (int e = 0; e < NUM_EXPERTS; ++e) l[e] = logits[t * NUM_EXPERTS + e];
    float m = l[0];
    #pragma unroll
    for (int e = 1; e < NUM_EXPERTS; ++e) m = fmaxf(m, l[e]);
    float s = 0.f;
    #pragma unroll
    for (int e = 0; e < NUM_EXPERTS; ++e) s += __expf(l[e] - m);
    float inv = 1.0f / s;

    int i1 = 0; float b1 = l[0];
    #pragma unroll
    for (int e = 1; e < NUM_EXPERTS; ++e) if (l[e] > b1) { b1 = l[e]; i1 = e; }
    int i2 = -1; float b2 = -3.4e38f;
    #pragma unroll
    for (int e = 0; e < NUM_EXPERTS; ++e)
        if (e != i1 && l[e] > b2) { b2 = l[e]; i2 = e; }

    int r1 = t * 2, r2 = t * 2 + 1;
    tw[r1] = __expf(b1 - m) * inv;
    tw[r2] = __expf(b2 - m) * inv;
    int p1 = atomicAdd(&cursor[i1], 1); rowmap[i1 * SEG + p1] = r1;
    int p2 = atomicAdd(&cursor[i2], 1); rowmap[i2 * SEG + p2] = r2;
}

// ---------------------------------------------------------------------
// Kernel 4: grouped GEMM. bf16 WMMA, f32 accumulate.
// Both operand tiles stream into LDS via GLOBAL_LOAD_ASYNC_TO_LDS_B64/B128
// (double-buffered, ASYNCcnt-paced, uniform 5 async ops per wave per tile);
// fused router-weight scale + top-k reduce via f32 global atomics.
// grid = (HIDDEN/BN, NUM_EXPERTS*TILES_PER_SEG), block = 256 (8 waves)
// ---------------------------------------------------------------------
__global__ __launch_bounds__(256) void moe_gemm(
    const unsigned short* __restrict__ xb,    // bf16 [NROWS, INTER]
    const unsigned short* __restrict__ wbT,   // bf16 [E, HIDDEN, INTER]
    const float* __restrict__ tw,
    const int*   __restrict__ rowmap,
    const int*   __restrict__ cnt,
    float*       __restrict__ out) {
    const int e    = blockIdx.y >> 9;            // expert
    const int tseg = blockIdx.y & (TILES_PER_SEG - 1);
    const int nrow = cnt[e];
    if (tseg * BM >= nrow) return;               // uniform early exit

    const int col0 = blockIdx.x * BN;
    const int tid  = threadIdx.x;

    __shared__ alignas(16) unsigned short sA[2][BM * KB];   // 2 x 2 KB  [row][k]
    __shared__ alignas(16) unsigned short sB[2][BN * KB];   // 2 x 16 KB [n][k]
    __shared__ float sTW[BM];
    __shared__ int   sRow[BM];

    if (tid < BM) {
        int slot = tseg * BM + tid;
        if (slot < nrow) {
            int r = rowmap[e * SEG + slot];
            sRow[tid] = r; sTW[tid] = tw[r];
        } else {
            sRow[tid] = 0; sTW[tid] = 0.0f;      // padded row contributes 0
        }
    }
    __syncthreads();

    const int wv = tid >> 5, lane = tid & 31;
    const int half = lane >> 4, lrow = lane & 15;

    // ---- async-copy addressing ----
    // B: 1024 x 16B chunks/tile; thread -> rows {q*64 + tid/4}, quarter tid&3
    const int trow = tid >> 2, bkq = tid & 3;
    const uint64_t gB0 = (uint64_t)(uintptr_t)wbT
        + (((size_t)e * HIDDEN + col0 + trow) * INTER) * 2 + (size_t)bkq * 16;
    const unsigned ldsB0 = (unsigned)(uintptr_t)(&sB[0][0]) + trow * 64 + bkq * 16;
    // A: 256 x 8B chunks/tile (uniform across all waves): row tid/8, eighth tid&7
    const int aro = tid >> 3, akq = tid & 7;
    const uint64_t gA0 = (uint64_t)(uintptr_t)xb
        + ((size_t)sRow[aro] * INTER) * 2 + (size_t)akq * 8;
    const unsigned ldsA0 = (unsigned)(uintptr_t)(&sA[0][0]) + aro * 64 + akq * 8;

    // exactly 5 async instructions per wave per tile -> uniform ASYNCcnt pacing
    auto issue = [&](int ks, int buf) {
        const unsigned koff = (unsigned)(ks * KB * 2);     // byte offset along K
        {
            uint64_t ga = gA0 + koff;
            unsigned ld = ldsA0 + (unsigned)buf * (BM * KB * 2);
            asm volatile("global_load_async_to_lds_b64 %0, %1, off"
                         :: "v"(ld), "v"(ga) : "memory");
        }
        #pragma unroll
        for (int q = 0; q < 4; ++q) {
            uint64_t ga = gB0 + koff + (unsigned)(q * 64 * INTER * 2);
            unsigned ld = ldsB0 + (unsigned)buf * (BN * KB * 2) + (unsigned)(q * 4096);
            asm volatile("global_load_async_to_lds_b128 %0, %1, off"
                         :: "v"(ld), "v"(ga) : "memory");
        }
    };

    v8f acc[2][2] = {};
    const int n0 = (wv * 2 + 0) * 16 + lrow;
    const int n1 = (wv * 2 + 1) * 16 + lrow;

    auto compute = [&](int buf) {
        union F { v16bf v; u32x4 u[2]; };
        F a0, a1, b0, b1;
        const unsigned short* pA = sA[buf];
        const unsigned short* pB = sB[buf];
        a0.u[0] = *(const u32x4*)&pA[(lrow     ) * KB + half * 8];
        a0.u[1] = *(const u32x4*)&pA[(lrow     ) * KB + 16 + half * 8];
        a1.u[0] = *(const u32x4*)&pA[(16 + lrow) * KB + half * 8];
        a1.u[1] = *(const u32x4*)&pA[(16 + lrow) * KB + 16 + half * 8];
        b0.u[0] = *(const u32x4*)&pB[n0 * KB + half * 8];
        b0.u[1] = *(const u32x4*)&pB[n0 * KB + 16 + half * 8];
        b1.u[0] = *(const u32x4*)&pB[n1 * KB + half * 8];
        b1.u[1] = *(const u32x4*)&pB[n1 * KB + 16 + half * 8];
        acc[0][0] = __builtin_amdgcn_wmma_f32_16x16x32_bf16(
                        false, a0.v, false, b0.v, (short)0, acc[0][0], false, false);
        acc[0][1] = __builtin_amdgcn_wmma_f32_16x16x32_bf16(
                        false, a0.v, false, b1.v, (short)0, acc[0][1], false, false);
        acc[1][0] = __builtin_amdgcn_wmma_f32_16x16x32_bf16(
                        false, a1.v, false, b0.v, (short)0, acc[1][0], false, false);
        acc[1][1] = __builtin_amdgcn_wmma_f32_16x16x32_bf16(
                        false, a1.v, false, b1.v, (short)0, acc[1][1], false, false);
    };

    // ---- software pipeline: 2 tiles in flight, branch-free steady state ----
    issue(0, 0);
    issue(1, 1);

    #pragma unroll 2
    for (int ks = 0; ks < NK - 2; ++ks) {
        asm volatile("s_wait_asynccnt 0x5" ::: "memory");  // tile ks resident
        __syncthreads();
        compute(ks & 1);
        __syncthreads();                                    // buffer free
        issue(ks + 2, ks & 1);
    }
    // ks = NK-2 (buf 0): next tile still in flight, nothing new to issue
    asm volatile("s_wait_asynccnt 0x5" ::: "memory");
    __syncthreads();
    compute(0);
    __syncthreads();
    // ks = NK-1 (buf 1): drain
    asm volatile("s_wait_asynccnt 0x0" ::: "memory");
    __syncthreads();
    compute(1);

    // ---- scale by router weight + fused top-k reduce (f32 global atomics) ----
    #pragma unroll
    for (int m = 0; m < 2; ++m) {
        #pragma unroll
        for (int j = 0; j < 8; ++j) {
            int row  = m * 16 + j + half * 8;     // C layout: VGPR j, half -> M
            float sc = sTW[row];
            int tok  = sRow[row] >> 1;            // r = t*TOPK + k -> token
            float* o = out + (size_t)tok * HIDDEN + col0;
            unsafeAtomicAdd(o + n0, acc[m][0][j] * sc);
            unsafeAtomicAdd(o + n1, acc[m][1][j] * sc);
        }
    }
}

// ---------------------------------------------------------------------
extern "C" void kernel_launch(void* const* d_in, const int* in_sizes, int n_in,
                              void* d_out, int out_size, void* d_ws, size_t ws_size,
                              hipStream_t stream) {
    const float* x      = (const float*)d_in[0];  // [NROWS, INTER]
    const float* w      = (const float*)d_in[1];  // [E, INTER, HIDDEN]
    const float* logits = (const float*)d_in[2];  // [NTOKENS, E]
    float* out = (float*)d_out;

    // workspace layout (needs ~68.2 MB)
    char* ws = (char*)d_ws;
    int*            cursor = (int*)ws;                              // 8 ints
    float*          tw     = (float*)(ws + 1024);                   // NROWS f32
    int*            rowmap = (int*)(ws + 1024 + (size_t)NROWS * 4); // E*SEG ints
    unsigned short* xb     = (unsigned short*)(ws + (1u << 20));    // 32 MB bf16
    unsigned short* wbT    = (unsigned short*)(ws + (1u << 20) + (size_t)NROWS * INTER * 2);

    hipMemsetAsync(cursor, 0, NUM_EXPERTS * sizeof(int), stream);
    hipMemsetAsync(out, 0, (size_t)NTOKENS * HIDDEN * sizeof(float), stream);

    xconv<<<(NROWS * (size_t)INTER) / (8 * 256), 256, 0, stream>>>(x, xb);
    wtrans<<<dim3(HIDDEN / 64, INTER / 64, NUM_EXPERTS), 256, 0, stream>>>(w, wbT);
    moe_route<<<NTOKENS / 256, 256, 0, stream>>>(logits, tw, rowmap, cursor);

    dim3 grid(HIDDEN / BN, NUM_EXPERTS * TILES_PER_SEG);
    moe_gemm<<<grid, 256, 0, stream>>>(xb, wbT, tw, rowmap, cursor, out);
}
